// Breadth_29248727286043
// MI455X (gfx1250) — matrix operand order repaired and checked by reading, not compiled, not run on hardware
//
#include <hip/hip_runtime.h>
#include <hip/hip_bf16.h>
#include <math.h>

#define HEADS 4
#define HEAD_DIM 32
#define DMODEL 128          // HEADS*HEAD_DIM == D_IN == 128
#define NEG_SLOPE 0.2f

typedef float    v2f  __attribute__((ext_vector_type(2)));
typedef float    v8f  __attribute__((ext_vector_type(8)));
typedef _Float16 v16h __attribute__((ext_vector_type(16)));

#if defined(__gfx1250__) && __has_builtin(__builtin_amdgcn_wmma_f32_16x16x4_f32)
#define USE_F32_WMMA 1
#else
#define USE_F32_WMMA 0
#endif

// -------------------- GEMM: h = x @ W  (WMMA, wave = 16x16 output tile) ----
// Block = 256 threads = 8 waves. Block b covers rows [16b,16b+16), wave w
// covers cols [16w,16w+16).  K = 128.  Full f32 precision (matches reference).
__global__ void gemm_wmma(const float* __restrict__ x,
                          const float* __restrict__ W,
                          float* __restrict__ h, int Nn) {
    const int lane = threadIdx.x & 31;
    const int wv   = threadIdx.x >> 5;      // 0..7 -> N-tile
    const int m    = lane & 15;             // row (A) / col (B,D) within tile
    const int hi   = lane >> 4;             // 0/1 -> K-pair select / D row half
    const int row0 = blockIdx.x * 16;
    const int col0 = wv * 16;

    int rA = row0 + m; if (rA >= Nn) rA = Nn - 1;       // clamp (keep EXEC full)
    const float* xrow = x + (size_t)rA * DMODEL;

    v8f acc = {};
#if USE_F32_WMMA
    // f32 WMMA: A 16x4 (2 f32/lane), B 4x16 (2 f32/lane), 32 MACs over K=128.
    for (int k0 = 0; k0 < DMODEL; k0 += 4) {
        const int kb = k0 + hi * 2;
        v2f a, b;
        a.x = xrow[kb];
        a.y = xrow[kb + 1];
        b.x = W[(size_t)(kb)     * DMODEL + col0 + m];
        b.y = W[(size_t)(kb + 1) * DMODEL + col0 + m];
        acc = __builtin_amdgcn_wmma_f32_16x16x4_f32(
            false, a, false, b, (short)0, acc, false, false);
    }
#else
    // Fallback: probe-confirmed f16 WMMA (f32 accumulate), K chunks of 32.
    for (int k0 = 0; k0 < DMODEL; k0 += 32) {
        v16h a, b;
        #pragma unroll
        for (int j = 0; j < 16; ++j) {
            const int kkA = (j < 8) ? (hi * 8 + j) : (16 + hi * 8 + (j - 8));
            a[j] = (_Float16)xrow[k0 + kkA];
            const int kkB = (j < 8) ? (hi * 8 + j) : (16 + hi * 8 + (j - 8));
            b[j] = (_Float16)W[(size_t)(k0 + kkB) * DMODEL + col0 + m];
        }
        acc = __builtin_amdgcn_wmma_f32_16x16x32_f16(
            false, a, false, b, (short)0, acc, false, false);
    }
#endif
    // D layout: vgpr v holds row = v + 8*hi, col = lane&15.
    // Uniform branch: only the last M-tile can be partial.
    float* hp = h + (size_t)(row0 + hi * 8) * DMODEL + col0 + m;
    if (row0 + 16 <= Nn) {
        #pragma unroll
        for (int v = 0; v < 8; ++v)
            hp[(size_t)v * DMODEL] = acc[v];
    } else {
        #pragma unroll
        for (int v = 0; v < 8; ++v)
            if (row0 + hi * 8 + v < Nn) hp[(size_t)v * DMODEL] = acc[v];
    }
}

// -------------------- per-node: attention dots + init ---------------------
__global__ void node_init(const float* __restrict__ h,
                          const float* __restrict__ att_src,
                          const float* __restrict__ att_dst,
                          float* __restrict__ asrc, float* __restrict__ adst,
                          float* __restrict__ mbuf, float* __restrict__ dnm,
                          float* __restrict__ out, int Nn) {
    const int t = blockIdx.x * blockDim.x + threadIdx.x;
    if (t >= Nn * HEADS) return;
    const int n  = t >> 2;
    const int hd = t & 3;
    const float* hp = h + (size_t)n * DMODEL + hd * HEAD_DIM;
    float s0 = 0.f, s1 = 0.f;
    #pragma unroll
    for (int i = 0; i < HEAD_DIM; ++i) {
        const float v = hp[i];
        s0 += v * att_src[hd * HEAD_DIM + i];
        s1 += v * att_dst[hd * HEAD_DIM + i];
    }
    asrc[t] = s0;
    adst[t] = s1;
    mbuf[t] = -3.402823466e38f;
    dnm[t]  = 0.f;
    float* op = out + (size_t)n * DMODEL + hd * HEAD_DIM;
    #pragma unroll
    for (int i = 0; i < HEAD_DIM; i += 4)
        *(float4*)(op + i) = make_float4(0.f, 0.f, 0.f, 0.f);
}

// monotone float max via signed-int max / unsigned-int min
__device__ __forceinline__ void atomicMaxFloat(float* addr, float v) {
    if (v >= 0.f) atomicMax((int*)addr, __float_as_int(v));
    else          atomicMin((unsigned int*)addr, __float_as_uint(v));
}

__device__ __forceinline__ void edge_pair(const int* ei, int E, int e, int& s, int& d) {
    if (e < E) { s = ei[e]; d = ei[E + e]; }         // edge_index is [2,E] row-major
    else       { s = d = e - E; }                    // self-loops appended
}

// -------------------- pass A: segment max of leaky logits -----------------
__global__ void edge_max(const int* __restrict__ ei, int E, int Nn,
                         const float* __restrict__ asrc,
                         const float* __restrict__ adst,
                         float* __restrict__ mbuf) {
    const int e = blockIdx.x * blockDim.x + threadIdx.x;
    if (e >= E + Nn) return;
    int s, d; edge_pair(ei, E, e, s, d);
    const float4 as = *(const float4*)(asrc + (size_t)s * 4);
    const float4 ad = *(const float4*)(adst + (size_t)d * 4);
    float lg[4] = {as.x + ad.x, as.y + ad.y, as.z + ad.z, as.w + ad.w};
    #pragma unroll
    for (int hh = 0; hh < 4; ++hh) {
        float l = lg[hh];
        l = (l >= 0.f) ? l : NEG_SLOPE * l;
        atomicMaxFloat(&mbuf[(size_t)d * 4 + hh], l);
    }
}

// -------------------- pass B: softmax denominator -------------------------
__global__ void edge_denom(const int* __restrict__ ei, int E, int Nn,
                           const float* __restrict__ asrc,
                           const float* __restrict__ adst,
                           const float* __restrict__ mbuf,
                           float* __restrict__ dnm) {
    const int e = blockIdx.x * blockDim.x + threadIdx.x;
    if (e >= E + Nn) return;
    int s, d; edge_pair(ei, E, e, s, d);
    const float4 as = *(const float4*)(asrc + (size_t)s * 4);
    const float4 ad = *(const float4*)(adst + (size_t)d * 4);
    const float4 mm = *(const float4*)(mbuf + (size_t)d * 4);
    float lg[4] = {as.x + ad.x, as.y + ad.y, as.z + ad.z, as.w + ad.w};
    float mv[4] = {mm.x, mm.y, mm.z, mm.w};
    #pragma unroll
    for (int hh = 0; hh < 4; ++hh) {
        float l = lg[hh];
        l = (l >= 0.f) ? l : NEG_SLOPE * l;
        atomicAdd(&dnm[(size_t)d * 4 + hh], expf(l - mv[hh]));
    }
}

// -------------------- pass C: weighted scatter (1 wave / edge) ------------
__global__ void edge_scatter(const int* __restrict__ ei, int E, int Nn,
                             const float* __restrict__ h,
                             const float* __restrict__ asrc,
                             const float* __restrict__ adst,
                             const float* __restrict__ mbuf,
                             const float* __restrict__ dnm,
                             float* __restrict__ out) {
    const int wave = (int)((blockIdx.x * blockDim.x + threadIdx.x) >> 5);
    const int lane = threadIdx.x & 31;
    if (wave >= E + Nn) return;
    int s, d; edge_pair(ei, E, wave, s, d);
    const int hd = lane >> 3;                         // 8 lanes (32 floats) per head
    float l = asrc[(size_t)s * 4 + hd] + adst[(size_t)d * 4 + hd];
    l = (l >= 0.f) ? l : NEG_SLOPE * l;
    const float alpha = expf(l - mbuf[(size_t)d * 4 + hd]) / dnm[(size_t)d * 4 + hd];
    const float4 hv = *(const float4*)(h + (size_t)s * DMODEL + lane * 4);
    float* op = out + (size_t)d * DMODEL + lane * 4;
    atomicAdd(op + 0, hv.x * alpha);
    atomicAdd(op + 1, hv.y * alpha);
    atomicAdd(op + 2, hv.z * alpha);
    atomicAdd(op + 3, hv.w * alpha);
}

// -------------------- finalize: bias + tanh -------------------------------
__global__ void finalize_k(float* __restrict__ out,
                           const float* __restrict__ bias, int total) {
    const int t = blockIdx.x * blockDim.x + threadIdx.x;
    if (t >= total) return;
    out[t] = tanhf(out[t] + bias[t & (DMODEL - 1)]);
}

extern "C" void kernel_launch(void* const* d_in, const int* in_sizes, int n_in,
                              void* d_out, int out_size, void* d_ws, size_t ws_size,
                              hipStream_t stream) {
    const float* x        = (const float*)d_in[0];
    const int*   ei       = (const int*)  d_in[1];   // [2,E] int32
    const float* W        = (const float*)d_in[2];
    const float* att_src  = (const float*)d_in[3];
    const float* att_dst  = (const float*)d_in[4];
    const float* bias     = (const float*)d_in[5];
    float*       out      = (float*)d_out;

    const int Nn = in_sizes[0] / DMODEL;             // 50000
    const int E  = in_sizes[1] / 2;                  // 600000
    const int Et = E + Nn;                           // + self loops

    // workspace layout (floats): h + a_src + a_dst + m + denom  (~28.8 MB)
    float* h    = (float*)d_ws;                      // N*128
    float* asrc = h    + (size_t)Nn * DMODEL;        // N*4
    float* adst = asrc + (size_t)Nn * HEADS;
    float* mbuf = adst + (size_t)Nn * HEADS;
    float* dnm  = mbuf + (size_t)Nn * HEADS;

    const int mtiles = (Nn + 15) / 16;
    gemm_wmma<<<mtiles, 256, 0, stream>>>(x, W, h, Nn);

    const int nthr = Nn * HEADS;
    node_init<<<(nthr + 255) / 256, 256, 0, stream>>>(h, att_src, att_dst,
                                                      asrc, adst, mbuf, dnm, out, Nn);

    edge_max  <<<(Et + 255) / 256, 256, 0, stream>>>(ei, E, Nn, asrc, adst, mbuf);
    edge_denom<<<(Et + 255) / 256, 256, 0, stream>>>(ei, E, Nn, asrc, adst, mbuf, dnm);

    const int waves_per_blk = 256 / 32;
    edge_scatter<<<(Et + waves_per_blk - 1) / waves_per_blk, 256, 0, stream>>>(
        ei, E, Nn, h, asrc, adst, mbuf, dnm, out);

    const int total = Nn * DMODEL;
    finalize_k<<<(total + 255) / 256, 256, 0, stream>>>(out, bias, total);
}
